// RNN_Build_74869869904702
// MI455X (gfx1250) — matrix operand order repaired and checked by reading, not compiled, not run on hardware
//
#include <hip/hip_runtime.h>
#include <math.h>

typedef __attribute__((ext_vector_type(16))) _Float16 v16h;
typedef __attribute__((ext_vector_type(8)))  _Float16 v8h;
typedef __attribute__((ext_vector_type(8)))  float    v8f;

#define EMBED   100
#define EPAD    128     // embedding columns padded to 128 (4 WMMA k-steps)
#define SEQ     80
#define UNITS   64
#define BATCH   4096
#define ROWS    32      // batch rows per block
#define THREADS 64      // 2 waves of 32
#define K1      192     // GEMM1 K: [x (128 padded) ; h1 (64)]
#define K2      128     // GEMM2 K: [h1 ; h2]

__device__ __forceinline__ float fast_tanh(float x) {
#if __has_builtin(__builtin_amdgcn_tanhf)
    return __builtin_amdgcn_tanhf(x);      // V_TANH_F32 hardware transcendental
#else
    return tanhf(x);
#endif
}

// A fragment (16-bit A, 16x32): lane holds row m = lane&15.
// lanes 0-15: K = kk*32 + {0..7, 16..23}; lanes 16-31: K = kk*32 + {8..15, 24..31}
__device__ __forceinline__ v16h load_a_frag(const _Float16* base, int stride, int kk, int lane) {
    const int m  = lane & 15;
    const int kb = kk * 32 + ((lane & 16) ? 8 : 0);
    const _Float16* p = base + m * stride + kb;
    v8h lo = *(const v8h*)(p);
    v8h hi = *(const v8h*)(p + 16);
    v16h a;
#pragma unroll
    for (int i = 0; i < 8; ++i) { a[i] = lo[i]; a[i + 8] = hi[i]; }
    return a;
}

// B fragment (16-bit B, 32x16): lane holds column n = lane&15.
// lanes 0-15: K = kk*32 + {0..7, 16..23} ; lanes 16-31: K = kk*32 + {8..15, 24..31}
// Weights stored transposed in LDS as [n][K] -> 2 contiguous b128 loads.
__device__ __forceinline__ v16h load_b_frag(const _Float16* wt, int stride, int ntile, int kk, int lane) {
    const int c  = ntile * 16 + (lane & 15);
    const int kb = kk * 32 + ((lane & 16) ? 8 : 0);
    const _Float16* p = wt + c * stride + kb;
    v8h lo = *(const v8h*)(p);
    v8h hi = *(const v8h*)(p + 16);
    v16h b;
#pragma unroll
    for (int i = 0; i < 8; ++i) { b[i] = lo[i]; b[i + 8] = hi[i]; }
    return b;
}

// CDNA5 async copy: 16B global -> LDS, per-lane addresses, tracked by ASYNCcnt.
__device__ __forceinline__ void async_copy16(unsigned lds_byte_off, const void* gaddr) {
    asm volatile("global_load_async_to_lds_b128 %0, %1, off"
                 :: "v"(lds_byte_off), "v"(gaddr) : "memory");
}
__device__ __forceinline__ void wait_async0() {
    asm volatile("s_wait_asynccnt 0" ::: "memory");
}

// One-time pre-pass: f32 embedding table -> f16, padded 100 -> 128 columns.
extern "C" __global__ void __launch_bounds__(256)
emb_to_f16(const float* __restrict__ emb, _Float16* __restrict__ embh, int nrows)
{
    int i = blockIdx.x * 256 + threadIdx.x;
    if (i < nrows * EPAD) {
        int r = i >> 7, c = i & 127;
        embh[i] = (_Float16)((c < EMBED) ? emb[r * EMBED + c] : 0.f);
    }
}

extern "C" __global__ void __launch_bounds__(THREADS)
rnn2_fused_wmma(const int* __restrict__ tokens,
                const _Float16* __restrict__ embh,
                const float* __restrict__ W1, const float* __restrict__ U1,
                const float* __restrict__ b1,
                const float* __restrict__ W2, const float* __restrict__ U2,
                const float* __restrict__ b2,
                const float* __restrict__ Wo, const float* __restrict__ bo,
                float* __restrict__ out)
{
    extern __shared__ _Float16 smem[];
    _Float16* WB1t = smem;                     // [UNITS][K1] = [W1;pad;U1]^T   24 KB
    _Float16* WB2t = WB1t + UNITS * K1;        // [UNITS][K2] = [W2;U2]^T       16 KB
    _Float16* Xbuf = WB2t + UNITS * K2;        // [2][ROWS][EPAD] double buffer 16 KB
    _Float16* A2   = Xbuf + 2 * ROWS * EPAD;   // [ROWS][K2]  = [h1 | h2]        8 KB

    const int tid  = threadIdx.x;
    const int lane = tid & 31;
    const int wave = tid >> 5;
    const int b0   = blockIdx.x * ROWS;

    // --- stage weights into LDS, transposed + f16 + zero-padded ---
    for (int i = tid; i < UNITS * K1; i += THREADS) {
        int n = i / K1, k = i - n * K1;
        float v = 0.f;
        if (k < EMBED)     v = W1[k * UNITS + n];
        else if (k >= 128) v = U1[(k - 128) * UNITS + n];
        WB1t[i] = (_Float16)v;
    }
    for (int i = tid; i < UNITS * K2; i += THREADS) {
        int n = i / K2, k = i - n * K2;
        float v = (k < UNITS) ? W2[k * UNITS + n] : U2[(k - UNITS) * UNITS + n];
        WB2t[i] = (_Float16)v;
    }
    for (int i = tid; i < ROWS * K2; i += THREADS) A2[i] = (_Float16)0.f;  // h1=h2=0

    const int lcol    = lane & 15;
    const int rowBase = wave * 16;
    const int rOff    = (lane & 16) ? 8 : 0;
    float b1v[4], b2v[4];
#pragma unroll
    for (int n = 0; n < 4; ++n) {
        b1v[n] = b1[n * 16 + lcol];
        b2v[n] = b2[n * 16 + lcol];
    }

    // per-thread gather assignment: row = tid/2, half-row (64 cols = 8 x 16B chunks)
    const int grow  = tid >> 1;
    const int ghalf = tid & 1;
    const unsigned xbufBase = (unsigned)(Xbuf - smem) * 2u;   // LDS byte offset of Xbuf

    // prologue: async-gather x for t=0 into buffer 0
    {
        int tok = tokens[(b0 + grow) * SEQ + 0];
        const _Float16* g = embh + (size_t)tok * EPAD + ghalf * 64;
        unsigned l = xbufBase + (unsigned)(grow * EPAD + ghalf * 64) * 2u;
#pragma unroll
        for (int j = 0; j < 8; ++j) async_copy16(l + j * 16, (const void*)(g + j * 8));
    }

    for (int t = 0; t < SEQ; ++t) {
        const _Float16* xb = Xbuf + (t & 1) * (ROWS * EPAD);

        wait_async0();       // my async writes into this step's buffer are done
        __syncthreads();     // everyone's are; everyone also finished reading other buffer

        // prefetch next timestep's x into the other buffer (fully overlapped with WMMA)
        if (t + 1 < SEQ) {
            int tok = tokens[(b0 + grow) * SEQ + (t + 1)];
            const _Float16* g = embh + (size_t)tok * EPAD + ghalf * 64;
            unsigned l = xbufBase +
                         (unsigned)(((t + 1) & 1) * ROWS * EPAD + grow * EPAD + ghalf * 64) * 2u;
#pragma unroll
            for (int j = 0; j < 8; ++j) async_copy16(l + j * 16, (const void*)(g + j * 8));
        }

        // --- GEMM1: h1 = tanh([x_t, h1] @ [W1;U1] + b1), K = 192 ---
        v8f acc[4];
#pragma unroll
        for (int n = 0; n < 4; ++n)
#pragma unroll
            for (int j = 0; j < 8; ++j) acc[n][j] = 0.f;

#pragma unroll
        for (int kk = 0; kk < 4; ++kk) {              // x part from double buffer
            v16h a = load_a_frag(xb + rowBase * EPAD, EPAD, kk, lane);
#pragma unroll
            for (int n = 0; n < 4; ++n) {
                v16h b = load_b_frag(WB1t, K1, n, kk, lane);
                acc[n] = __builtin_amdgcn_wmma_f32_16x16x32_f16(
                    false, a, false, b, (short)0, acc[n], false, false);
            }
        }
#pragma unroll
        for (int kk = 4; kk < 6; ++kk) {              // U1*h1 part: h1 lives in A2[:,0:64]
            v16h a = load_a_frag(A2 + rowBase * K2, K2, kk - 4, lane);
#pragma unroll
            for (int n = 0; n < 4; ++n) {
                v16h b = load_b_frag(WB1t, K1, n, kk, lane);
                acc[n] = __builtin_amdgcn_wmma_f32_16x16x32_f16(
                    false, a, false, b, (short)0, acc[n], false, false);
            }
        }
#pragma unroll
        for (int n = 0; n < 4; ++n) {
            int col = n * 16 + lcol;
#pragma unroll
            for (int r = 0; r < 8; ++r) {
                _Float16 hv = (_Float16)fast_tanh(acc[n][r] + b1v[n]);
                A2[(rowBase + r + rOff) * K2 + col] = hv;
            }
        }

        // --- GEMM2: h2 = tanh([h1, h2] @ [W2;U2] + b2), K = 128 ---
        v8f acc2[4];
#pragma unroll
        for (int n = 0; n < 4; ++n)
#pragma unroll
            for (int j = 0; j < 8; ++j) acc2[n][j] = 0.f;

#pragma unroll
        for (int kk = 0; kk < 4; ++kk) {
            v16h a = load_a_frag(A2 + rowBase * K2, K2, kk, lane);
#pragma unroll
            for (int n = 0; n < 4; ++n) {
                v16h b = load_b_frag(WB2t, K2, n, kk, lane);
                acc2[n] = __builtin_amdgcn_wmma_f32_16x16x32_f16(
                    false, a, false, b, (short)0, acc2[n], false, false);
            }
        }
#pragma unroll
        for (int n = 0; n < 4; ++n) {
            int col = n * 16 + lcol;
#pragma unroll
            for (int r = 0; r < 8; ++r) {
                _Float16 hv = (_Float16)fast_tanh(acc2[n][r] + b2v[n]);
                A2[(rowBase + r + rOff) * K2 + 64 + col] = hv;
            }
        }
    }
    __syncthreads();

    // --- head: out = sigmoid(h2 @ Wo + bo) ---
    if (tid < ROWS) {
        float s = bo[0];
#pragma unroll 8
        for (int k = 0; k < UNITS; ++k)
            s += (float)A2[tid * K2 + 64 + k] * Wo[k];
        out[b0 + tid] = 1.f / (1.f + expf(-s));
    }
}

extern "C" void kernel_launch(void* const* d_in, const int* in_sizes, int n_in,
                              void* d_out, int out_size, void* d_ws, size_t ws_size,
                              hipStream_t stream) {
    (void)in_sizes; (void)n_in; (void)ws_size; (void)out_size;
    const int*   tokens = (const int*)  d_in[0];
    const float* emb    = (const float*)d_in[1];
    const float* W1     = (const float*)d_in[2];
    const float* U1     = (const float*)d_in[3];
    const float* b1     = (const float*)d_in[4];
    const float* W2     = (const float*)d_in[5];
    const float* U2     = (const float*)d_in[6];
    const float* b2     = (const float*)d_in[7];
    const float* Wo     = (const float*)d_in[8];
    const float* bo     = (const float*)d_in[9];
    float* out = (float*)d_out;

    const int TOTAL_WORDS = 10000;
    _Float16* embh = (_Float16*)d_ws;   // needs 10000*128*2 = 2.56 MB of scratch

    dim3 cgrid((TOTAL_WORDS * EPAD + 255) / 256), cblk(256);
    emb_to_f16<<<cgrid, cblk, 0, stream>>>(emb, embh, TOTAL_WORDS);

    const size_t shmem = (size_t)(UNITS * K1 + UNITS * K2 + 2 * ROWS * EPAD + ROWS * K2) * 2u;
    dim3 grid(BATCH / ROWS), block(THREADS);
    rnn2_fused_wmma<<<grid, block, shmem, stream>>>(tokens, embh, W1, U1, b1,
                                                    W2, U2, b2, Wo, bo, out);
}
